// Decoder_16449724744065
// MI455X (gfx1250) — matrix operand order repaired and checked by reading, not compiled, not run on hardware
//
#include <hip/hip_runtime.h>
#include <hip/hip_bf16.h>
#include <math.h>

typedef __attribute__((ext_vector_type(16))) _Float16 v16h;
typedef __attribute__((ext_vector_type(8)))  _Float16 v8h;
typedef __attribute__((ext_vector_type(4)))  _Float16 v4h;
typedef __attribute__((ext_vector_type(8)))  float    v8f;

#define Bn 4096
#define Tn 28
#define Fn 127
#define An 3072
#define Hn 128
#define Gn 384          // 3*H
#define K1n 3199        // IN minus prev_y column
#define KPn 3200        // padded K (multiple of 32)
#define NCn 100         // KPn / 32 k-chunks
#define Mn (Bn*Tn)      // 114688

__device__ __forceinline__ v8f wmma_f16(v16h a, v16h b, v8f c) {
  // D = A(16x32 f16) * B(32x16 f16) + C(16x16 f32)
  return __builtin_amdgcn_wmma_f32_16x16x32_f16(false, a, false, b, (short)0, c,
                                                false, false);
}

// ---------------------------------------------------------------------------
// Pack W_ih columns 1..3199 (+1 zero pad col) into B-fragment-major f16:
// layout [nt=0..23][c=0..99][lane=0..31][half=0..15]
// element: N = nt*16 + (L&15), K = c*32 + (L>>4)*16 + h
// ---------------------------------------------------------------------------
__global__ __launch_bounds__(256) void pack_wih(const float* __restrict__ Wih,
                                                _Float16* __restrict__ W16B) {
  int idx = blockIdx.x * 256 + threadIdx.x;      // < 24*100*512
  int h = idx & 15, L = (idx >> 4) & 31;
  int frag = idx >> 9;
  int c = frag % NCn, nt = frag / NCn;
  int N = nt * 16 + (L & 15);
  int K = c * 32 + ((L >> 4) * 16) + h;
  float v = (K < K1n) ? Wih[(size_t)N * KPn + K + 1] : 0.0f;
  W16B[idx] = (_Float16)v;
}

// Pack W_hh (384x128) into B-fragment-major f16: [ct=0..23][c=0..3][lane][half]
__global__ __launch_bounds__(256) void pack_whh(const float* __restrict__ Whh,
                                                _Float16* __restrict__ Whh16B) {
  int idx = blockIdx.x * 256 + threadIdx.x;      // < 24*4*512 = 49152
  int h = idx & 15, L = (idx >> 4) & 31;
  int c = (idx >> 9) & 3, ct = idx >> 11;
  int j = ct * 16 + (L & 15);
  int k = c * 32 + ((L >> 4) * 16) + h;
  Whh16B[idx] = (_Float16)Whh[j * Hn + k];
}

// ---------------------------------------------------------------------------
// Big precompute GEMM:  GI[m][j] = sum_k X[m][k] * Wpack[j][k]
//   m = t*B + b, X row = [feat(127) | attn(3072) | pad0], K=3200, N=384.
// Block: 512 threads = 16 waves; 64 M-rows per block.
// Wave (g=w>>2, q=w&3) computes rows g*16..+15 x col tiles q*6..q*6+5.
// Manually ping-ponged pipeline: B frags double-buffered in registers
// (B0/B1, no rotation copies), A chunks double-buffered in LDS.
// ---------------------------------------------------------------------------
__global__ __launch_bounds__(512) void gemm_gi(const float* __restrict__ pred,
                                               const float* __restrict__ attn,
                                               const _Float16* __restrict__ W16B,
                                               float* __restrict__ GI) {
  __shared__ _Float16 lA[2][4 * 32 * 16];  // 2 x (4 row-groups x 32 x 16 halves)
  const int tid = threadIdx.x;
  const int blk = blockIdx.x;              // 64 rows each
  // ---- staging role: thread -> (local row, 4-K subchunk) ----
  const int sr = tid >> 3;                 // 0..63
  const int c4 = tid & 7;                  // 0..7
  const int m  = blk * 64 + sr;
  const int t  = m >> 12;                  // m / 4096
  const int b  = m & 4095;
  const float* predRow = pred + ((size_t)b * Tn + t) * Fn;
  const float* attnRow = attn + ((size_t)b * Tn + t) * An;
  const int sg    = sr >> 4;
  const int slane = (sr & 15) + ((c4 >> 2) << 4);
  const int shoff = 4 * (c4 & 3);
  const int sidx  = (sg * 32 + slane) * 16 + shoff;
  // ---- compute role ----
  const int w = tid >> 5;
  const int L = tid & 31;
  const int g = w >> 2;                    // 16-row group 0..3
  const int q = w & 3;                     // N quarter 0..3 (tiles q*6..q*6+5)
  const _Float16* lAsrc = &lA[0][(g * 32 + L) * 16];

  v8f acc[6];
  #pragma unroll
  for (int i = 0; i < 6; ++i) acc[i] = (v8f){};

  auto stageX = [&](int chunk, int buf) {
    const int k0 = chunk * 32 + c4 * 4;
    v4h tmp;
    if (chunk >= 4 && chunk < NCn - 1) {
      // branch-free fast path: window fully inside attn
      const float* p = attnRow + (k0 - Fn);
      #pragma unroll
      for (int i = 0; i < 4; ++i) tmp[i] = (_Float16)p[i];
      __builtin_prefetch(p + 64, 0, 1);    // 2 k-chunks ahead (HBM stream)
    } else {
      #pragma unroll
      for (int i = 0; i < 4; ++i) {
        int k = k0 + i;
        float v = (k < Fn) ? predRow[k] : ((k < K1n) ? attnRow[k - Fn] : 0.0f);
        tmp[i] = (_Float16)v;
      }
    }
    *(v4h*)&lA[buf][sidx] = tmp;
  };
  auto loadB = [&](int chunk, v16h* dst) {
    #pragma unroll
    for (int i = 0; i < 6; ++i)
      dst[i] = *(const v16h*)(W16B +
                  ((size_t)((q * 6 + i) * NCn + chunk) * 32 + L) * 16);
  };
  auto runWmma = [&](const v16h* Bx, int buf) {
    v16h a = *(const v16h*)(lAsrc + buf * (4 * 32 * 16));
    #pragma unroll
    for (int i = 0; i < 6; ++i) acc[i] = wmma_f16(a, Bx[i], acc[i]);
  };

  v16h B0[6], B1[6];
  loadB(0, B0);
  stageX(0, 0);
  __syncthreads();

  for (int c = 0; c < NCn; c += 2) {
    // even phase: prefetch chunk c+1 (regs + LDS buf1), compute chunk c
    loadB(c + 1, B1);
    stageX(c + 1, 1);
    runWmma(B0, 0);
    __syncthreads();
    // odd phase: prefetch chunk c+2 (regs + LDS buf0), compute chunk c+1
    if (c + 2 < NCn) {
      loadB(c + 2, B0);
      stageX(c + 2, 0);
    }
    runWmma(B1, 1);
    __syncthreads();
  }

  // epilogue: C frag (VGPR v, lane L) -> row v+8*(L>>4), col L&15
  const int hi = L >> 4;
  #pragma unroll
  for (int i = 0; i < 6; ++i) {
    int j = (q * 6 + i) * 16 + (L & 15);
    #pragma unroll
    for (int v = 0; v < 8; ++v) {
      int row = g * 16 + v + 8 * hi;
      GI[(size_t)(blk * 64 + row) * Gn + j] = acc[i][v];
    }
  }
}

// ---------------------------------------------------------------------------
// Fused 28-step GRU recurrence. Batch rows are independent, so each block
// owns 16 rows and runs all T steps with only block-local barriers.
// Block: 8 waves; wave nt owns u-strip [nt*16, nt*16+16); its three C tiles
// (col tiles nt, nt+8, nt+16) are the r/z/n gates of the SAME (b,u) element
// per lane slot. W_hh B-fragments (12 x v16h) live in registers all 28 steps.
// h is kept twice in LDS: f32 (hT, for z*h and projection) and f16 (hF, in
// u-contiguous layout so each A fragment rebuild is 2 x ds_load_b128).
// ---------------------------------------------------------------------------
#define HFS 136   // hF row stride in halves (bank-conflict padding)

__global__ __launch_bounds__(256) void gru_steps(
    const float* __restrict__ GI, const _Float16* __restrict__ Whh16B,
    const float* __restrict__ enc, const float* __restrict__ prevy,
    const float* __restrict__ Wih, const float* __restrict__ bih,
    const float* __restrict__ bhh, const float* __restrict__ Wproj,
    const float* __restrict__ bproj, float* __restrict__ out) {
  __shared__ float     hT[16 * 129];   // f32 h, stride 129 (bank-safe)
  __shared__ _Float16  hF[16 * HFS];   // f16 h, row stride 136 halves
  __shared__ float     py[16];
  __shared__ float     sWp[128];
  const int tid = threadIdx.x;
  const int r0  = blockIdx.x * 16;
  const int w   = tid >> 5;        // wave = u strip index nt
  const int L   = tid & 31;

  #pragma unroll
  for (int i = 0; i < 8; ++i) {    // load initial hidden state (coalesced)
    int idx = tid + i * 256;
    int row = idx >> 7, u = idx & 127;
    float hv = enc[(size_t)(r0 + row) * Hn + u];
    hT[row * 129 + u] = hv;
    hF[row * HFS + u] = (_Float16)hv;
  }
  if (tid < 16)  py[tid]  = prevy[r0 + tid];
  if (tid < 128) sWp[tid] = Wproj[tid];

  // W_hh B-fragments: col tiles {w, w+8, w+16}, K chunks 0..3 -> registers
  v16h Bf[12];
  #pragma unroll
  for (int qg = 0; qg < 3; ++qg) {
    int ct = w + qg * 8;
    #pragma unroll
    for (int c = 0; c < 4; ++c)
      Bf[qg * 4 + c] = *(const v16h*)(Whh16B + ((ct * 4 + c) * 32 + L) * 16);
  }
  // per-lane gate constants (j_r = u = w*16 + (L&15), fixed all steps)
  const int jr = w * 16 + (L & 15);
  const float bihr = bih[jr], bihz = bih[jr + 128], bihn = bih[jr + 256];
  const float bhhr = bhh[jr], bhhz = bhh[jr + 128], bhhn = bhh[jr + 256];
  const float w0r = Wih[(size_t)jr * KPn];
  const float w0z = Wih[(size_t)(jr + 128) * KPn];
  const float w0n = Wih[(size_t)(jr + 256) * KPn];
  const float bp  = bproj[0];
  const int   hi  = L >> 4;
  const int   arow = L & 15;               // A-frag row for this lane
  __syncthreads();

  for (int t = 0; t < Tn; ++t) {
    // build A fragments: two contiguous 8-half runs per chunk -> 2 b128 loads
    v16h Af[4];
    #pragma unroll
    for (int c = 0; c < 4; ++c) {
      int u0 = c * 32 + (L >> 4) * 16;
      v8h lo = *(const v8h*)&hF[arow * HFS + u0];
      v8h hh = *(const v8h*)&hF[arow * HFS + u0 + 8];
      Af[c] = __builtin_shufflevector(lo, hh, 0, 1, 2, 3, 4, 5, 6, 7,
                                      8, 9, 10, 11, 12, 13, 14, 15);
    }
    // gh = h @ W_hh^T for the three gate strips (K=128 -> 4 WMMA each)
    v8f gr = (v8f){}, gz = (v8f){}, gn = (v8f){};
    #pragma unroll
    for (int c = 0; c < 4; ++c) {
      gr = wmma_f16(Af[c], Bf[0 + c], gr);
      gz = wmma_f16(Af[c], Bf[4 + c], gz);
      gn = wmma_f16(Af[c], Bf[8 + c], gn);
    }
    // elementwise gates; C slot v <-> batch row v+8*hi, col jr
    float hnew[8];
    size_t gibase = ((size_t)t * Bn + r0) * Gn;
    #pragma unroll
    for (int v = 0; v < 8; ++v) {
      int row = v + 8 * hi;
      float pyv = py[row];
      size_t gm = gibase + (size_t)row * Gn;
      float gir = GI[gm + jr]       + bihr + pyv * w0r;
      float giz = GI[gm + jr + 128] + bihz + pyv * w0z;
      float gin = GI[gm + jr + 256] + bihn + pyv * w0n;
      float ghr = gr[v] + bhhr;
      float ghz = gz[v] + bhhz;
      float ghn = gn[v] + bhhn;
      float rg = 1.0f / (1.0f + expf(-(gir + ghr)));
      float zg = 1.0f / (1.0f + expf(-(giz + ghz)));
      float ng = tanhf(gin + rg * ghn);
      hnew[v] = (1.0f - zg) * ng + zg * hT[row * 129 + jr];
    }
    __syncthreads();                          // all hT/hF reads of step t done
    #pragma unroll
    for (int v = 0; v < 8; ++v) {
      int row = v + 8 * hi;
      hT[row * 129 + jr] = hnew[v];
      hF[row * HFS + jr] = (_Float16)hnew[v];
      out[(size_t)Tn * Bn + ((size_t)t * Bn + r0 + row) * Hn + jr] = hnew[v];
    }
    __syncthreads();                          // hT/hF hold h_{t+1}
    if (tid < 16) {                           // projection: y = h_new . Wproj
      float a = bp;
      for (int u = 0; u < 128; ++u) a += hT[tid * 129 + u] * sWp[u];
      out[(size_t)t * Bn + r0 + tid] = a;     // targets[t][b]
      py[tid] = a;                            // feeds next step
    }
    __syncthreads();
  }
}

extern "C" void kernel_launch(void* const* d_in, const int* in_sizes, int n_in,
                              void* d_out, int out_size, void* d_ws, size_t ws_size,
                              hipStream_t stream) {
  const float* enc   = (const float*)d_in[0];
  const float* attn  = (const float*)d_in[1];
  const float* pred  = (const float*)d_in[2];
  const float* prevy = (const float*)d_in[3];
  const float* Wih   = (const float*)d_in[4];
  const float* Whh   = (const float*)d_in[5];
  const float* bih   = (const float*)d_in[6];
  const float* bhh   = (const float*)d_in[7];
  const float* Wproj = (const float*)d_in[8];
  const float* bproj = (const float*)d_in[9];
  float* out = (float*)d_out;

  // workspace layout
  _Float16* W16B   = (_Float16*)d_ws;                             // 2,457,600 B
  _Float16* Whh16B = (_Float16*)((char*)d_ws + 2457600);          //    98,304 B
  float*    GI     = (float*)((char*)d_ws + 2555904);             // 176,160,768 B

  pack_wih <<<4800, 256, 0, stream>>>(Wih, W16B);
  pack_whh <<<192,  256, 0, stream>>>(Whh, Whh16B);
  gemm_gi  <<<Mn / 64, 512, 0, stream>>>(pred, attn, W16B, GI);
  gru_steps<<<Bn / 16, 256, 0, stream>>>(GI, Whh16B, enc, prevy, Wih,
                                         bih, bhh, Wproj, bproj, out);
}
